// ReactionMapper_76106820485631
// MI455X (gfx1250) — compile-verified, matrix operand order checked by reading, not run on hardware
//
#include <hip/hip_runtime.h>
#include <hip/hip_bf16.h>
#include <math.h>

// ============================================================================
// Set-Transformer hypergraph encoder for MI455X (gfx1250, wave32, WMMA).
//
//  * rxn_idx = arange(E)//8  ->  dense [R, 8, 128] gather, mask all-true.
//  * All heavy math is x[16x128] @ W^T[128x128]: v_wmma_f32_16x16x32_f16,
//    2 reactions packed per wave (16 rows = WMMA M).
//  * Weights pre-converted once (prep kernel) to f16 WMMA-B tiles in d_ws; the
//    main kernel preloads all 4 K-step tiles per N-tile into registers before
//    the WMMA chain so L2 latency overlaps the matrix pipe.
// ============================================================================

typedef __attribute__((ext_vector_type(16))) _Float16 v16h;
typedef __attribute__((ext_vector_type(8)))  float    v8f;
typedef __attribute__((ext_vector_type(4)))  float    v4f;

#define CC     128
#define HEADS  4
#define HD     32
#define SS     8
#define RR     131072
#define EE     (RR * SS)
#define TILE_HALFS (128 * 128)   // halfs per prepped 128x128 weight matrix
#define NSLOT  18                // number of prepped weight matrices

// ---------------------------------------------------------------------------
// Input index map. setup_inputs() dict order: metabolite_features,
// hyperedge_index, params.  params flattened JAX-tree style (sorted keys):
//  dec0{in_b,in_w,lin_b,lin_w,ln1_b,ln1_g,ln2_b,ln2_g,out_b,out_w},
//  enc0{...}, enc1{...}, pma_lin_b, pma_lin_w, pma_mab{...}, seed
// ---------------------------------------------------------------------------
enum {
  IN_FEATS = 0, IN_HE = 1,
  D_IN_B = 2, D_IN_W, D_LIN_B, D_LIN_W, D_LN1B, D_LN1G, D_LN2B, D_LN2G, D_OUT_B, D_OUT_W,
  E0_IN_B = 12, E0_IN_W, E0_LIN_B, E0_LIN_W, E0_LN1B, E0_LN1G, E0_LN2B, E0_LN2G, E0_OUT_B, E0_OUT_W,
  E1_IN_B = 22, E1_IN_W, E1_LIN_B, E1_LIN_W, E1_LN1B, E1_LN1G, E1_LN2B, E1_LN2G, E1_OUT_B, E1_OUT_W,
  P_LIN_B = 32, P_LIN_W = 33,
  PM_IN_B = 34, PM_IN_W, PM_LIN_B, PM_LIN_W, PM_LN1B, PM_LN1G, PM_LN2B, PM_LN2G, PM_OUT_B, PM_OUT_W,
  IN_SEED = 44
};

// Prepped-weight slot layout inside d_ws (f16):
//  0..4  enc0: q,k,v,out,lin    5..9  enc1: q,k,v,out,lin
//  10 pma_lin   11 pma_k  12 pma_v  13 pma_out  14 pma_ffn
//  15 dec_v     16 dec_out  17 dec_ffn
// followed by qseed[128] (f32) at byte offset NSLOT*TILE_HALFS*2.

struct MabP { const float *in_b, *out_b, *lin_b, *g1, *b1, *g2, *b2; };

struct MainP {
  const float* feats; const int* he;
  const _Float16* wt; const float* qseed;
  const float* pma_lin_b; const float* seedvec;
  MabP e0, e1, pm, de;
  float* out;
};

struct PrepP { const float* src[NSLOT]; _Float16* wt; };

// ---------------------------------------------------------------------------
// Prep kernel: scatter each W[N=128][K=128] (row-major f32) into the WMMA-B
// f16 tile layout.  Tile (nt,kt) lane L half h holds B[k][n] = W[n][k] with
//   n = nt*16 + (L&15),  k = kt*32 + (L>>4)*8 + (h&7) + (h>>3)*16
// (mirrors the 16-bit A-operand striping with M<->N swapped).
// ---------------------------------------------------------------------------
__global__ void prep_weights(PrepP P) {
  int idx = blockIdx.x * blockDim.x + threadIdx.x;
  if (idx >= NSLOT * TILE_HALFS) return;
  int slot = idx / TILE_HALFS;
  int r    = idx % TILE_HALFS;
  int h    = r & 15;
  int lane = (r >> 4) & 31;
  int t    = r >> 9;               // t = nt*4 + kt
  int nt = t >> 2, kt = t & 3;
  int n = nt * 16 + (lane & 15);
  int k = kt * 32 + ((lane >> 4) & 1) * 8 + (h & 7) + (h >> 3) * 16;
  P.wt[idx] = (_Float16)P.src[slot][n * CC + k];
}

// qseed[n] = seed . Wq[n,:] + bq[n]  (identical for every reaction)
__global__ void prep_qseed(const float* seed, const float* inw_q,
                           const float* inb_q, float* qseed) {
  int n = threadIdx.x;
  float a = 0.f;
  for (int k = 0; k < CC; ++k) a += seed[k] * inw_q[n * CC + k];
  qseed[n] = a + inb_q[n];
}

// ---------------------------------------------------------------------------
// Device helpers (single wave per workgroup; __syncthreads is a NOP/fence).
// ---------------------------------------------------------------------------
__device__ inline v16h load_btile(const _Float16* wt, int nt, int kt, int lane) {
  return *(const v16h*)(wt + ((nt * 4 + kt) * 32 + lane) * 16);
}

// Build the 4 A-operands (16x32 f16 each) for one 16x128 activation tile.
__device__ inline void build_a4(const float* src, int lane, v16h a[4]) {
  int m = lane & 15, hi = (lane >> 4) & 1;
#pragma unroll
  for (int kt = 0; kt < 4; ++kt) {
    const float* p = src + m * CC + kt * 32 + hi * 8;
    const v4f* p4 = (const v4f*)p;
    v4f x0 = p4[0], x1 = p4[1], x2 = p4[4], x3 = p4[5];  // k..k+7, k+16..k+23
    v16h x;
#pragma unroll
    for (int j = 0; j < 4; ++j) { x[j]     = (_Float16)x0[j]; x[4 + j]  = (_Float16)x1[j]; }
#pragma unroll
    for (int j = 0; j < 4; ++j) { x[8 + j] = (_Float16)x2[j]; x[12 + j] = (_Float16)x3[j]; }
    a[kt] = x;
  }
}

// dst[16x128] = act( src[16x128] @ W^T + bias ), W as prepped f16 tiles.
// All 4 B tiles of an N-tile are fetched into registers before the WMMA chain
// so the loads issue as one clause and overlap the matrix pipe.
template <bool RELU>
__device__ void gemm128(float* dst, const float* src, const _Float16* wt,
                        const float* bias, int lane) {
  v16h a[4];
  build_a4(src, lane, a);
  int col = lane & 15, hi = (lane >> 4) & 1;
#pragma unroll
  for (int nt = 0; nt < 8; ++nt) {
    v16h b[4];
#pragma unroll
    for (int kt = 0; kt < 4; ++kt) b[kt] = load_btile(wt, nt, kt, lane);
    float bv = bias[nt * 16 + col];
    v8f acc = {};
#pragma unroll
    for (int kt = 0; kt < 4; ++kt) {
      acc = __builtin_amdgcn_wmma_f32_16x16x32_f16(
          /*neg_a=*/false, a[kt], /*neg_b=*/false, b[kt],
          /*c_mod=*/(short)0, acc, /*reuse_a=*/false, /*reuse_b=*/false);
    }
#pragma unroll
    for (int p = 0; p < 8; ++p) {
      float v = acc[p] + bv;
      if (RELU) v = fmaxf(v, 0.f);
      dst[(p + 8 * hi) * CC + nt * 16 + col] = v;
    }
  }
}

__device__ inline void add_inplace(float* x, const float* a, int lane) {
  for (int i = lane * 4; i < 16 * CC; i += 32 * 4) {
    v4f* px = (v4f*)(x + i);
    const v4f* pa = (const v4f*)(a + i);
    v4f v = *px, w = *pa;
#pragma unroll
    for (int e = 0; e < 4; ++e) v[e] += w[e];
    *px = v;
  }
}

// Row-wise LayerNorm over 128.  lane L: row = L&15, cols (L>>4)*64..+63;
// cross-half reduction via shfl_xor(16).
__device__ void layernorm(float* x, const float* g, const float* b, int lane) {
  int row = lane & 15, hi = (lane >> 4) & 1;
  float* p = x + row * CC + hi * 64;
  float s = 0.f;
  for (int j = 0; j < 64; ++j) s += p[j];
  s += __shfl_xor(s, 16, 32);
  float mu = s * (1.f / 128.f);
  float var = 0.f;
  for (int j = 0; j < 64; ++j) { float d = p[j] - mu; var += d * d; }
  var += __shfl_xor(var, 16, 32);
  float inv = rsqrtf(var * (1.f / 128.f) + 1e-5f);
  const float* gg = g + hi * 64;
  const float* bb = b + hi * 64;
  for (int j = 0; j < 64; ++j) p[j] = (p[j] - mu) * inv * gg[j] + bb[j];
}

// 8-key attention (2 reactions stacked: rows 0-7 / 8-15).
// lane L: q-row t = L&15, heads (L>>4)*2 .. +1.
__device__ void attn8(const float* q, const float* k, const float* v,
                      float* o, int lane) {
  int t = lane & 15, hb = ((lane >> 4) & 1) * 2;
  int kbase = (t >> 3) * 8;
  for (int h = hb; h < hb + 2; ++h) {
    const float* qp = q + t * CC + h * HD;
    float sc[8], mx = -1e30f;
    for (int j = 0; j < 8; ++j) {
      const float* kp = k + (kbase + j) * CC + h * HD;
      float d = 0.f;
      for (int c = 0; c < HD; ++c) d += qp[c] * kp[c];
      sc[j] = d * 0.17677669529663687f;  // 1/sqrt(32)
      mx = fmaxf(mx, sc[j]);
    }
    float den = 0.f;
    for (int j = 0; j < 8; ++j) { sc[j] = __expf(sc[j] - mx); den += sc[j]; }
    float rden = 1.f / den;
    float acc[HD];
#pragma unroll
    for (int c = 0; c < HD; ++c) acc[c] = 0.f;
    for (int j = 0; j < 8; ++j) {
      const v4f* vp = (const v4f*)(v + (kbase + j) * CC + h * HD);
      float a = sc[j];
#pragma unroll
      for (int q4 = 0; q4 < HD / 4; ++q4) {
        v4f vv = vp[q4];
#pragma unroll
        for (int e = 0; e < 4; ++e) acc[q4 * 4 + e] += a * vv[e];
      }
    }
    float* op = o + t * CC + h * HD;
#pragma unroll
    for (int c = 0; c < HD; ++c) op[c] = acc[c] * rden;
  }
}

__device__ void sab_enc(float* sX, float* sQ, float* sK, float* sV, float* sT,
                        const _Float16* wt5, const MabP& mp, int lane) {
  gemm128<false>(sQ, sX, wt5 + 0 * TILE_HALFS, mp.in_b + 0 * CC, lane);
  gemm128<false>(sK, sX, wt5 + 1 * TILE_HALFS, mp.in_b + 1 * CC, lane);
  gemm128<false>(sV, sX, wt5 + 2 * TILE_HALFS, mp.in_b + 2 * CC, lane);
  __syncthreads();
  attn8(sQ, sK, sV, sT, lane);
  __syncthreads();
  gemm128<false>(sQ, sT, wt5 + 3 * TILE_HALFS, mp.out_b, lane);  // out proj
  __syncthreads();
  add_inplace(sX, sQ, lane);
  __syncthreads();
  layernorm(sX, mp.g1, mp.b1, lane);
  __syncthreads();
  gemm128<true>(sT, sX, wt5 + 4 * TILE_HALFS, mp.lin_b, lane);   // FFN
  __syncthreads();
  add_inplace(sX, sT, lane);
  __syncthreads();
  layernorm(sX, mp.g2, mp.b2, lane);
  __syncthreads();
}

// ---------------------------------------------------------------------------
// Main kernel: 1 wave / workgroup, 2 reactions / wave.
// ---------------------------------------------------------------------------
__global__ __launch_bounds__(32) void setformer_main(MainP P) {
  __shared__ float sX[16 * CC], sQ[16 * CC], sK[16 * CC], sV[16 * CC], sT[16 * CC];
  const int lane = threadIdx.x;
  const int r0 = blockIdx.x * 2;

  __builtin_prefetch(P.wt, 0, 0);  // global_prefetch_b8 for first weight tiles

  // Gather: rows 0-7 <- reaction r0, rows 8-15 <- reaction r0+1 (mask all-true)
  // 16 rows x 32 float4 per row; one global_load_b128 + ds_store_b128 each.
  for (int i = lane; i < 16 * 32; i += 32) {
    int t = i >> 5, c4 = i & 31;
    int e = (r0 + (t >> 3)) * SS + (t & 7);
    int met = P.he[e];  // hyperedge_index row 0 (met indices)
    const v4f* src = (const v4f*)(P.feats + met * CC);
    ((v4f*)sX)[t * 32 + c4] = src[c4];
  }
  __syncthreads();

  // --- two encoder SABs ---
  sab_enc(sX, sQ, sK, sV, sT, P.wt + 0 * TILE_HALFS, P.e0, lane);
  sab_enc(sX, sQ, sK, sV, sT, P.wt + 5 * TILE_HALFS, P.e1, lane);

  // --- PMA (1 seed point) ---
  gemm128<true>(sT, sX, P.wt + 10 * TILE_HALFS, P.pma_lin_b, lane);  // h = relu(lin)
  __syncthreads();
  gemm128<false>(sK, sT, P.wt + 11 * TILE_HALFS, P.pm.in_b + 1 * CC, lane);
  gemm128<false>(sV, sT, P.wt + 12 * TILE_HALFS, P.pm.in_b + 2 * CC, lane);
  __syncthreads();
  for (int i = lane; i < 16 * CC; i += 32) sQ[i] = 0.f;
  __syncthreads();
  if (lane < 8) {  // (reaction, head) pairs; q is the precomputed seed query
    int r = lane >> 2, h = lane & 3;
    const float* qp = P.qseed + h * HD;
    float sc[8], mx = -1e30f;
    for (int j = 0; j < 8; ++j) {
      const float* kp = sK + (r * 8 + j) * CC + h * HD;
      float d = 0.f;
      for (int c = 0; c < HD; ++c) d += qp[c] * kp[c];
      sc[j] = d * 0.17677669529663687f;
      mx = fmaxf(mx, sc[j]);
    }
    float den = 0.f;
    for (int j = 0; j < 8; ++j) { sc[j] = __expf(sc[j] - mx); den += sc[j]; }
    float rden = 1.f / den;
    for (int c = 0; c < HD; ++c) {
      float acc = 0.f;
      for (int j = 0; j < 8; ++j) acc += sc[j] * sV[(r * 8 + j) * CC + h * HD + c];
      sQ[r * CC + h * HD + c] = acc * rden;
    }
  }
  __syncthreads();
  gemm128<false>(sT, sQ, P.wt + 13 * TILE_HALFS, P.pm.out_b, lane);  // out proj
  __syncthreads();
  for (int i = lane; i < 16 * CC; i += 32) {  // residual = seed; pad rows zero
    int t = i >> 7, c = i & 127;
    sX[i] = (t < 2) ? (P.seedvec[c] + sT[i]) : 0.f;
  }
  __syncthreads();
  layernorm(sX, P.pm.g1, P.pm.b1, lane);
  __syncthreads();
  gemm128<true>(sT, sX, P.wt + 14 * TILE_HALFS, P.pm.lin_b, lane);
  __syncthreads();
  add_inplace(sX, sT, lane);
  __syncthreads();
  layernorm(sX, P.pm.g2, P.pm.b2, lane);
  __syncthreads();

  // --- decoder SAB on [1,128]: softmax over 1 key => MHA output == V proj ---
  gemm128<false>(sV, sX, P.wt + 15 * TILE_HALFS, P.de.in_b + 2 * CC, lane);
  __syncthreads();
  gemm128<false>(sT, sV, P.wt + 16 * TILE_HALFS, P.de.out_b, lane);
  __syncthreads();
  add_inplace(sX, sT, lane);
  __syncthreads();
  layernorm(sX, P.de.g1, P.de.b1, lane);
  __syncthreads();
  gemm128<true>(sT, sX, P.wt + 17 * TILE_HALFS, P.de.lin_b, lane);
  __syncthreads();
  add_inplace(sX, sT, lane);
  __syncthreads();
  layernorm(sX, P.de.g2, P.de.b2, lane);
  __syncthreads();

  // --- write rows 0,1 with nan_to_num semantics ---
  for (int i = lane; i < 2 * CC; i += 32) {
    int t = i >> 7, c = i & 127;
    float v = sX[t * CC + c];
    if (v != v) v = 0.f;
    v = fminf(fmaxf(v, -3.4028234663852886e38f), 3.4028234663852886e38f);
    P.out[(r0 + t) * CC + c] = v;
  }
}

// ---------------------------------------------------------------------------
extern "C" void kernel_launch(void* const* d_in, const int* in_sizes, int n_in,
                              void* d_out, int out_size, void* d_ws, size_t ws_size,
                              hipStream_t stream) {
  (void)in_sizes; (void)n_in; (void)out_size; (void)ws_size;
  auto f = [&](int i) { return (const float*)d_in[i]; };

  _Float16* wt = (_Float16*)d_ws;
  float* qseed = (float*)((char*)d_ws + (size_t)NSLOT * TILE_HALFS * sizeof(_Float16));

  // --- prep: f32 -> f16 WMMA-B tiles ---
  PrepP pp;
  pp.wt = wt;
  pp.src[0]  = f(E0_IN_W) + 0;      pp.src[1]  = f(E0_IN_W) + 16384;
  pp.src[2]  = f(E0_IN_W) + 32768;  pp.src[3]  = f(E0_OUT_W);
  pp.src[4]  = f(E0_LIN_W);
  pp.src[5]  = f(E1_IN_W) + 0;      pp.src[6]  = f(E1_IN_W) + 16384;
  pp.src[7]  = f(E1_IN_W) + 32768;  pp.src[8]  = f(E1_OUT_W);
  pp.src[9]  = f(E1_LIN_W);
  pp.src[10] = f(P_LIN_W);
  pp.src[11] = f(PM_IN_W) + 16384;  pp.src[12] = f(PM_IN_W) + 32768;
  pp.src[13] = f(PM_OUT_W);         pp.src[14] = f(PM_LIN_W);
  pp.src[15] = f(D_IN_W) + 32768;   pp.src[16] = f(D_OUT_W);
  pp.src[17] = f(D_LIN_W);
  {
    int total = NSLOT * TILE_HALFS;
    prep_weights<<<(total + 255) / 256, 256, 0, stream>>>(pp);
  }
  prep_qseed<<<1, 128, 0, stream>>>(f(IN_SEED), f(PM_IN_W), f(PM_IN_B), qseed);

  // --- main ---
  MainP mp;
  mp.feats = f(IN_FEATS);
  mp.he = (const int*)d_in[IN_HE];
  mp.wt = wt;
  mp.qseed = qseed;
  mp.pma_lin_b = f(P_LIN_B);
  mp.seedvec = f(IN_SEED);
  mp.e0 = { f(E0_IN_B), f(E0_OUT_B), f(E0_LIN_B), f(E0_LN1G), f(E0_LN1B), f(E0_LN2G), f(E0_LN2B) };
  mp.e1 = { f(E1_IN_B), f(E1_OUT_B), f(E1_LIN_B), f(E1_LN1G), f(E1_LN1B), f(E1_LN2G), f(E1_LN2B) };
  mp.pm = { f(PM_IN_B), f(PM_OUT_B), f(PM_LIN_B), f(PM_LN1G), f(PM_LN1B), f(PM_LN2G), f(PM_LN2B) };
  mp.de = { f(D_IN_B),  f(D_OUT_B),  f(D_LIN_B),  f(D_LN1G),  f(D_LN1B),  f(D_LN2G),  f(D_LN2B)  };
  mp.out = (float*)d_out;

  setformer_main<<<RR / 2, 32, 0, stream>>>(mp);
}
